// LSTM_1589137900073
// MI455X (gfx1250) — compile-verified, matrix operand order checked by reading, not run on hardware
//
#include <hip/hip_runtime.h>
#include <hip/hip_bf16.h>

// Problem constants (match reference)
#define B_  256
#define S_  512
#define F_  64
#define H_  1024
#define O_  24
#define G4H (4 * H_)   // 4096

#define TILE_M    128
#define CHUNK_K   32
#define LDS_ROW_B 80   // 64 data bytes + 16 pad: 16B-aligned, bank-conflict-free

typedef __attribute__((ext_vector_type(8)))  __bf16 v8bf;
typedef __attribute__((ext_vector_type(16))) __bf16 v16bf;
typedef __attribute__((ext_vector_type(8)))  float  v8f;

static __device__ __forceinline__ unsigned short f2bf_rne(float x) {
  unsigned int u = __float_as_uint(x);
  u += 0x7FFFu + ((u >> 16) & 1u);   // round-to-nearest-even
  return (unsigned short)(u >> 16);
}

// ---- CDNA5 async copy: global -> LDS, 16 bytes per lane, tracked by ASYNCcnt ----
static __device__ __forceinline__ void async_load_b128(const void* gptr, void* lptr) {
  // Low 32 bits of a generic pointer to LDS == LDS byte offset (aperture rule).
  unsigned lds_off = (unsigned)(unsigned long long)lptr;
  asm volatile("global_load_async_to_lds_b128 %0, %1, off"
               :: "v"(lds_off), "v"(gptr) : "memory");
}

static __device__ __forceinline__ void wait_asynccnt0() {
#if defined(__has_builtin) && __has_builtin(__builtin_amdgcn_s_wait_asynccnt)
  __builtin_amdgcn_s_wait_asynccnt(0);
#else
  asm volatile("s_wait_asynccnt 0x0" ::: "memory");
#endif
}

// Load one bf16 WMMA fragment row-chunk straight from global, per CDNA5 ISA layout:
// lanes 0-15 hold K {k..k+7, k+16..k+23}; lanes 16-31 hold K {k+8..k+15, k+24..k+31}.
// Caller passes p = row_base + k + ((lane>>4)*8); we read 8 elems at p and 8 at p+16.
static __device__ __forceinline__ v16bf load_frag(const __bf16* p) {
  v8bf lo = *(const v8bf*)p;
  v8bf hi = *(const v8bf*)(p + 16);
  v16bf r;
#pragma unroll
  for (int i = 0; i < 8; ++i) { r[i] = lo[i]; r[i + 8] = hi[i]; }
  return r;
}

// Same fragment but from an LDS-staged row (row stride LDS_ROW_B bytes).
static __device__ __forceinline__ v16bf load_frag_lds(const char* p) {
  v8bf lo = *(const v8bf*)p;          // 16B, ds_load_b128
  v8bf hi = *(const v8bf*)(p + 32);   // +16 bf16 elements
  v16bf r;
#pragma unroll
  for (int i = 0; i < 8; ++i) { r[i] = lo[i]; r[i + 8] = hi[i]; }
  return r;
}

// Issue async copies staging A[mtile .. mtile+127][kbase .. kbase+31] into an LDS buffer.
// 512 x b128 chunks spread over 256 threads (2 each).
static __device__ __forceinline__ void stage_async(
    const __bf16* __restrict__ A, int lda, int kbase, int mtile,
    char* smbuf, int tid)
{
#pragma unroll
  for (int it = 0; it < 2; ++it) {
    const int c   = tid + it * 256;
    const int row = c >> 2;        // 0..127
    const int q   = c & 3;         // 16B quarter within the 64B row chunk
    const __bf16* g = A + (size_t)(mtile + row) * lda + kbase + q * 8;
    async_load_b128(g, smbuf + row * LDS_ROW_B + q * 16);
  }
}

// One K-segment of the gates GEMM with LDS double buffering: acc += A_seg @ W_seg^T
// NOTE: Kseg/32 must be EVEN so stage(0) of the next segment never targets the
// buffer read in the immediately-preceding iteration (true here: 64->2, 1024->32).
static __device__ __forceinline__ void gemm_segment(
    const __bf16* __restrict__ A, int lda, int Kseg, int mtile,
    const __bf16* __restrict__ W, int ldw, int wkoff,
    char* sm0, char* sm1,
    int n0, int rA, int khalf, int lrow_base, int tid,
    v8f acc[4][2])
{
  const int nk = Kseg / CHUNK_K;
  stage_async(A, lda, 0, mtile, sm0, tid);
  for (int ki = 0; ki < nk; ++ki) {
    char* cur = (ki & 1) ? sm1 : sm0;
    char* nxt = (ki & 1) ? sm0 : sm1;
    wait_asynccnt0();        // my async stores to LDS complete
    __syncthreads();         // everyone's complete -> tile visible; prev reads fenced
    if (ki + 1 < nk) stage_async(A, lda, (ki + 1) * CHUNK_K, mtile, nxt, tid);

    const int k = ki * CHUNK_K;
    v16bf b[2];
#pragma unroll
    for (int j = 0; j < 2; ++j) {
      const __bf16* p = W + (size_t)(n0 + j * 16 + rA) * ldw + wkoff + k + khalf;
      b[j] = load_frag(p);
      if (k + CHUNK_K < Kseg) __builtin_prefetch(p + CHUNK_K, 0, 3); // global_prefetch_b8
    }
    v16bf a[4];
#pragma unroll
    for (int i = 0; i < 4; ++i) {
      const char* p = cur + (lrow_base + i * 16 + rA) * LDS_ROW_B + khalf * 2;
      a[i] = load_frag_lds(p);
    }
#pragma unroll
    for (int i = 0; i < 4; ++i)
#pragma unroll
      for (int j = 0; j < 2; ++j)
        acc[i][j] = __builtin_amdgcn_wmma_f32_16x16x32_bf16(
            false, a[i], false, b[j], (short)0, acc[i][j], false, false);
  }
}

// gates[256, 4096] = [x_t | h_prev] @ [W_ih | W_hh]^T + bias   (bf16 in, f32 out)
// grid = 64 blocks (2 M-tiles x 32 N-tiles of 128x128), 256 threads = 8 waves.
// A tiles staged via async-to-LDS double buffering; B (weights) direct from L2.
__global__ __launch_bounds__(256) void gates_wmma(
    const unsigned short* __restrict__ A0u, int lda0, int K0,
    const unsigned short* __restrict__ A1u, int lda1, int K1,
    const unsigned short* __restrict__ Wu,  int ldw,
    const float* __restrict__ bias,
    float* __restrict__ gates)
{
  __shared__ __align__(16) char smem[2][TILE_M * LDS_ROW_B];

  const __bf16* A0 = (const __bf16*)A0u;
  const __bf16* A1 = (const __bf16*)A1u;
  const __bf16* W  = (const __bf16*)Wu;

  const int tid    = threadIdx.x;
  const int lane   = tid & 31;
  const int wave   = tid >> 5;
  const int wave_m = wave & 1;        // 0..1  -> 64 rows each
  const int wave_n = wave >> 1;       // 0..3  -> 32 cols each
  const int tm     = blockIdx.x & 1;  // 2 M tiles of 128
  const int tn     = blockIdx.x >> 1; // 32 N tiles of 128

  const int khalf  = (lane >> 4) << 3; // 0 or 8
  const int rA     = lane & 15;
  const int mtile  = tm * TILE_M;
  const int lrow0  = wave_m * 64;          // wave's M base within the LDS tile
  const int n0     = tn * 128 + wave_n * 32;

  v8f acc[4][2];
  v8f vz;
#pragma unroll
  for (int q = 0; q < 8; ++q) vz[q] = 0.0f;
#pragma unroll
  for (int i = 0; i < 4; ++i)
#pragma unroll
    for (int j = 0; j < 2; ++j) acc[i][j] = vz;

  gemm_segment(A0, lda0, K0, mtile, W, ldw, 0,  smem[0], smem[1],
               n0, rA, khalf, lrow0, tid, acc);
  gemm_segment(A1, lda1, K1, mtile, W, ldw, K0, smem[0], smem[1],
               n0, rA, khalf, lrow0, tid, acc);

  // C layout: lanes 0-15: M = base+v, N = n0+lane; lanes 16-31: M = base+8+v, N = lane-16.
#pragma unroll
  for (int i = 0; i < 4; ++i)
#pragma unroll
    for (int j = 0; j < 2; ++j) {
      const int mb = mtile + lrow0 + i * 16 + ((lane >> 4) << 3);
      const int n  = n0 + j * 16 + (lane & 15);
      const float bn = bias[n];
#pragma unroll
      for (int v = 0; v < 8; ++v)
        gates[(size_t)(mb + v) * G4H + n] = acc[i][j][v] + bn;
    }
}

// Elementwise LSTM cell. B*H threads. Writes new c (f32), h (bf16 for next GEMM),
// optionally h (f32) for the fc accumulation (layer 1 only).
__global__ __launch_bounds__(256) void lstm_cell(
    const float* __restrict__ gates, float* __restrict__ c,
    unsigned short* __restrict__ h_bf, float* __restrict__ h_f32)
{
  const int idx = blockIdx.x * blockDim.x + threadIdx.x;
  const int b = idx >> 10;          // / H_
  const int h = idx & (H_ - 1);
  const float* g = gates + (size_t)b * G4H;
  const float gi = g[h];
  const float gf = g[H_ + h];
  const float gg = g[2 * H_ + h];
  const float go = g[3 * H_ + h];
  const float si = 1.0f / (1.0f + __expf(-gi));
  const float sf = 1.0f / (1.0f + __expf(-gf));
  const float so = 1.0f / (1.0f + __expf(-go));
  const float cn = sf * c[idx] + si * tanhf(gg);
  const float hn = so * tanhf(cn);
  c[idx]    = cn;
  h_bf[idx] = f2bf_rne(hn);
  if (h_f32) h_f32[idx] = hn;
}

// Per-step fc rank-update: acc[b,o] += h2_t[b,:] . fc_W[o, t*H : t*H+H]
// grid = B blocks of 768 threads = 24 waves; wave w owns output o = w.
__global__ __launch_bounds__(768) void fc_step(
    const float* __restrict__ h2, const float* __restrict__ fcW, int t,
    float* __restrict__ acc)
{
  const int b = blockIdx.x;
  const int o = threadIdx.x >> 5;   // 0..23
  const int lane = threadIdx.x & 31;
  const float* hrow = h2 + (size_t)b * H_;
  const float* wrow = fcW + (size_t)o * ((size_t)S_ * H_) + (size_t)t * H_;
  float s = 0.0f;
#pragma unroll 8
  for (int h = lane; h < H_; h += 32) s += hrow[h] * wrow[h];
#pragma unroll
  for (int off = 16; off > 0; off >>= 1) s += __shfl_down(s, off, 32);
  if (lane == 0) acc[b * O_ + o] += s;
}

// Pack [W_ih | W_hh] row-major (4096 x (Fin+H)) as bf16.
__global__ void pack_weights(const float* __restrict__ Wih,
                             const float* __restrict__ Whh,
                             int Fin, unsigned short* __restrict__ Wcat)
{
  const int ld = Fin + H_;
  const long long total = (long long)G4H * ld;
  const long long idx = (long long)blockIdx.x * blockDim.x + threadIdx.x;
  if (idx >= total) return;
  const int row = (int)(idx / ld);
  const int col = (int)(idx % ld);
  const float v = (col < Fin) ? Wih[(size_t)row * Fin + col]
                              : Whh[(size_t)row * H_ + (col - Fin)];
  Wcat[idx] = f2bf_rne(v);
}

__global__ void make_bias(const float* __restrict__ bih,
                          const float* __restrict__ bhh,
                          float* __restrict__ bias)
{
  const int i = blockIdx.x * blockDim.x + threadIdx.x;
  if (i < G4H) bias[i] = bih[i] + bhh[i];
}

// x [B,S,F] f32 -> xT [S,B,F] bf16
__global__ void convert_x(const float* __restrict__ x,
                          unsigned short* __restrict__ xT)
{
  const long long idx = (long long)blockIdx.x * blockDim.x + threadIdx.x;
  if (idx >= (long long)S_ * B_ * F_) return;
  const int f = (int)(idx % F_);
  const long long sb = idx / F_;
  const int b = (int)(sb % B_);
  const int s = (int)(sb / B_);
  xT[idx] = f2bf_rne(x[((size_t)b * S_ + s) * F_ + f]);
}

__global__ void zero_u32(unsigned int* __restrict__ p, long long n) {
  const long long i = (long long)blockIdx.x * blockDim.x + threadIdx.x;
  if (i < n) p[i] = 0u;
}

__global__ void add_bias_out(const float* __restrict__ acc,
                             const float* __restrict__ fcb,
                             float* __restrict__ out)
{
  const int i = blockIdx.x * blockDim.x + threadIdx.x;
  if (i < B_ * O_) out[i] = acc[i] + fcb[i % O_];
}

static inline int cdiv_ll(long long a, long long b) { return (int)((a + b - 1) / b); }

extern "C" void kernel_launch(void* const* d_in, const int* in_sizes, int n_in,
                              void* d_out, int out_size, void* d_ws, size_t ws_size,
                              hipStream_t stream) {
  (void)in_sizes; (void)n_in; (void)out_size; (void)ws_size;
  const float* x    = (const float*)d_in[0];
  const float* Wih0 = (const float*)d_in[1];
  const float* Whh0 = (const float*)d_in[2];
  const float* bih0 = (const float*)d_in[3];
  const float* bhh0 = (const float*)d_in[4];
  const float* Wih1 = (const float*)d_in[5];
  const float* Whh1 = (const float*)d_in[6];
  const float* bih1 = (const float*)d_in[7];
  const float* bhh1 = (const float*)d_in[8];
  const float* fcW  = (const float*)d_in[9];
  const float* fcb  = (const float*)d_in[10];
  float* out = (float*)d_out;

  // ---- workspace carve-out (~51 MB total) ----
  char* ws = (char*)d_ws;
  size_t off = 0;
  auto carve = [&](size_t bytes) -> char* {
    char* p = ws + off;
    off = (off + bytes + 255) & ~(size_t)255;
    return p;
  };
  unsigned short* Wcat0 = (unsigned short*)carve((size_t)G4H * (F_ + H_) * 2);
  unsigned short* Wcat1 = (unsigned short*)carve((size_t)G4H * (H_ + H_) * 2);
  float* bias0 = (float*)carve((size_t)G4H * 4);
  float* bias1 = (float*)carve((size_t)G4H * 4);
  unsigned short* xT = (unsigned short*)carve((size_t)S_ * B_ * F_ * 2);
  float* gbuf = (float*)carve((size_t)B_ * G4H * 4);
  const size_t zoff0 = off;                        // everything below is zero-inited
  unsigned short* h0 = (unsigned short*)carve((size_t)B_ * H_ * 2);
  float* c0 = (float*)carve((size_t)B_ * H_ * 4);
  unsigned short* h1 = (unsigned short*)carve((size_t)B_ * H_ * 2);
  float* c1 = (float*)carve((size_t)B_ * H_ * 4);
  float* h1f = (float*)carve((size_t)B_ * H_ * 4);
  float* accO = (float*)carve((size_t)B_ * O_ * 4);
  const size_t zbytes = off - zoff0;

  // ---- setup ----
  pack_weights<<<cdiv_ll((long long)G4H * (F_ + H_), 256), 256, 0, stream>>>(Wih0, Whh0, F_, Wcat0);
  pack_weights<<<cdiv_ll((long long)G4H * (H_ + H_), 256), 256, 0, stream>>>(Wih1, Whh1, H_, Wcat1);
  make_bias<<<G4H / 256, 256, 0, stream>>>(bih0, bhh0, bias0);
  make_bias<<<G4H / 256, 256, 0, stream>>>(bih1, bhh1, bias1);
  convert_x<<<cdiv_ll((long long)S_ * B_ * F_, 256), 256, 0, stream>>>(x, xT);
  zero_u32<<<cdiv_ll((long long)(zbytes / 4), 256), 256, 0, stream>>>(
      (unsigned int*)(ws + zoff0), (long long)(zbytes / 4));

  // ---- sequential time loop: 2 LSTM layers + streaming fc update ----
  for (int t = 0; t < S_; ++t) {
    const unsigned short* xt = xT + (size_t)t * B_ * F_;
    // layer 0: gates = [x_t | h0] @ [Wih0|Whh0]^T + b
    gates_wmma<<<64, 256, 0, stream>>>(xt, F_, F_, h0, H_, H_, Wcat0, F_ + H_, bias0, gbuf);
    lstm_cell<<<(B_ * H_) / 256, 256, 0, stream>>>(gbuf, c0, h0, nullptr);
    // layer 1: gates = [h0(t) | h1] @ [Wih1|Whh1]^T + b
    gates_wmma<<<64, 256, 0, stream>>>(h0, H_, H_, h1, H_, H_, Wcat1, H_ + H_, bias1, gbuf);
    lstm_cell<<<(B_ * H_) / 256, 256, 0, stream>>>(gbuf, c1, h1, h1f);
    // fc: accO += h2_t @ fc_W[:, t*H:(t+1)*H]^T
    fc_step<<<B_, 768, 0, stream>>>(h1f, fcW, t, accO);
  }
  add_bias_out<<<cdiv_ll(B_ * O_, 256), 256, 0, stream>>>(accO, fcb, out);
}
